// Seq2Seq_16088947491583
// MI455X (gfx1250) — matrix operand order repaired and checked by reading, not compiled, not run on hardware
//
#include <hip/hip_runtime.h>
#include <hip/hip_bf16.h>

// ---------------------------------------------------------------------------
// Seq2Seq GRU (H=256, L=32, 20 decode steps, batch=1) for gfx1250 (MI455X).
// Single persistent workgroup (8 wave32s) on one WGP; weights converted once
// to bf16 in WMMA A-fragment layout (pre-warms L2), matvecs done with
// v_wmma_f32_16x16x32_bf16 (B columns replicated with the activation vector).
// B fragments are hoisted into VGPRs once per step per wave so each WMMA
// chain only waits on the clause-batched global A-fragment loads.
// ---------------------------------------------------------------------------

typedef __attribute__((ext_vector_type(16))) __bf16 v16bf;
typedef __attribute__((ext_vector_type(8)))  float  v8f;

#define H 256
#define L 32
#define STEPS 20
#define V_OUT 74
#define BOS 2
#define EOS 3

#define GATE_TILES 48      // 768 / 16
#define KCHUNKS 8          // 256 / 32
#define FC_TILES 5         // 80 / 16 (74 padded to 80)
#define TILE_ELEMS (KCHUNKS * 32 * 16)   // bf16 elems per output tile (4096)
#define TILES_PER_WAVE 12  // 48 tiles / 4 waves per matrix

static __device__ inline __bf16 f32_to_bf16(float f) {
    unsigned u = __builtin_bit_cast(unsigned, f);
    u += 0x7FFFu + ((u >> 16) & 1u);                // round-to-nearest-even
    unsigned short h = (unsigned short)(u >> 16);
    return __builtin_bit_cast(__bf16, h);
}

static __device__ inline float sigmoidf_(float x) {
    return 1.0f / (1.0f + __expf(-x));
}

// ---------------------------------------------------------------------------
// Weight conversion: fp32 row-major [src_rows, 256] -> bf16, laid out exactly
// as the per-lane v16bf A-fragment stream for V_WMMA_*_16X16X32_BF16.
// dst flat index d = (((tile*8 + kc)*32 + lane)*16 + j)
//   lane<16 : M = lane,     K = kc*32 + j
//   lane>=16: M = lane-16,  K = kc*32 + 16 + j
// Rows beyond src_rows (fc padding 74->80) are zero-filled.
// ---------------------------------------------------------------------------
__global__ void convert_weights_k(const float* __restrict__ src,
                                  __bf16* __restrict__ dst,
                                  int src_rows, int total) {
    int d = blockIdx.x * blockDim.x + threadIdx.x;
    if (d >= total) return;
    int j    =  d        & 15;
    int lane = (d >> 4)  & 31;
    int kc   = (d >> 9)  & 7;
    int t    =  d >> 12;
    int row  = t * 16 + (lane & 15);
    int jj   = (lane < 16) ? j
                           : j + 16;
    int col  = kc * 32 + jj;
    float v  = (row < src_rows) ? src[row * H + col] : 0.0f;
    dst[d] = f32_to_bf16(v);
}

// ---------------------------------------------------------------------------
// ntiles consecutive 16-output tiles of a [rows x 256] * [256] matvec on the
// matrix pipe. The 8 replicated B fragments (the activation vector, shared by
// every tile) are hoisted into VGPRs once; each tile is then a pure chain of
// global A-fragment loads + 8 WMMAs.
// Result per tile: lanes with (lane&15)==0 hold the 16 outputs split 8/8
// across the two lane halves in the 8 accumulator VGPRs.
// ---------------------------------------------------------------------------
static __device__ inline void matvec_tiles(const __bf16* __restrict__ Wfrag,
                                           const __bf16* vec_lds,
                                           float* out_lds,
                                           int tile0, int ntiles, int lane) {
    v16bf b[KCHUNKS];
#pragma unroll
    for (int kc = 0; kc < KCHUNKS; ++kc)
        b[kc] = *(const v16bf*)(vec_lds + kc * 32 + ((lane & 16) ? 16 : 0));

    const v16bf* wp = (const v16bf*)Wfrag;
#pragma unroll 2
    for (int t = tile0; t < tile0 + ntiles; ++t) {
        v8f c = {};
#pragma unroll
        for (int kc = 0; kc < KCHUNKS; ++kc) {
            v16bf a = wp[(t * KCHUNKS + kc) * 32 + lane];
            c = __builtin_amdgcn_wmma_f32_16x16x32_bf16(
                    /*neg_a=*/false, a, /*neg_b=*/false, b[kc],
                    /*c_mod=*/(short)0, c, /*reuse_a=*/false, /*reuse_b=*/false);
        }
        if ((lane & 15) == 0) {
            float* dst = out_lds + t * 16 + ((lane >> 4) * 8);
#pragma unroll
            for (int r = 0; r < 8; ++r) dst[r] = c[r];
        }
    }
}

// ---------------------------------------------------------------------------
// Persistent kernel: full encoder + greedy decoder in one workgroup.
// ---------------------------------------------------------------------------
__global__ void __launch_bounds__(256, 1)
seq2seq_main_k(const int*   __restrict__ enc_input,
               const float* __restrict__ enc_emb,
               const float* __restrict__ enc_bih,
               const float* __restrict__ enc_bhh,
               const float* __restrict__ dec_emb,
               const float* __restrict__ dec_bih,
               const float* __restrict__ dec_bhh,
               const float* __restrict__ fc_b,
               const __bf16* __restrict__ encWih,
               const __bf16* __restrict__ encWhh,
               const __bf16* __restrict__ decWih,
               const __bf16* __restrict__ decWhh,
               const __bf16* __restrict__ fcW,
               float* __restrict__ out) {
    __shared__ float sh_gi[3 * H];
    __shared__ float sh_gh[3 * H];
    __shared__ float sh_h[H];
    __shared__ __align__(32) __bf16 sh_hbf[H];
    __shared__ __align__(32) __bf16 sh_xbf[H];
    __shared__ float sh_logits[FC_TILES * 16];
    __shared__ int sh_tok;
    __shared__ int sh_done;

    const int tid  = threadIdx.x;
    const int wave = tid >> 5;
    const int lane = tid & 31;

    sh_h[tid]   = 0.0f;
    sh_hbf[tid] = f32_to_bf16(0.0f);
    if (tid == 0) { sh_tok = BOS; sh_done = 0; }
    __syncthreads();

    // ---------------- encoder: 32 sequential GRU steps ----------------
    for (int step = 0; step < L; ++step) {
        int tok = enc_input[step];
        sh_xbf[tid] = f32_to_bf16(enc_emb[tok * H + tid]);
        __syncthreads();

        // waves 0-3: gi = Wih*x (48 tiles); waves 4-7: gh = Whh*h (48 tiles)
        {
            const __bf16* W   = (wave < 4) ? encWih  : encWhh;
            const __bf16* vec = (wave < 4) ? sh_xbf  : sh_hbf;
            float*        og  = (wave < 4) ? sh_gi   : sh_gh;
            matvec_tiles(W, vec, og, (wave & 3) * TILES_PER_WAVE,
                         TILES_PER_WAVE, lane);
        }
        __syncthreads();

        // gate math: one thread per hidden unit
        {
            int i = tid;
            float ir = sh_gi[i]         + enc_bih[i];
            float iz = sh_gi[H + i]     + enc_bih[H + i];
            float in = sh_gi[2 * H + i] + enc_bih[2 * H + i];
            float hr = sh_gh[i]         + enc_bhh[i];
            float hz = sh_gh[H + i]     + enc_bhh[H + i];
            float hn = sh_gh[2 * H + i] + enc_bhh[2 * H + i];
            float r  = sigmoidf_(ir + hr);
            float z  = sigmoidf_(iz + hz);
            float n  = tanhf(in + r * hn);
            float hv = (1.0f - z) * n + z * sh_h[i];
            sh_h[i]   = hv;
            sh_hbf[i] = f32_to_bf16(hv);
        }
        __syncthreads();
    }

    // ---------------- decoder: 20 greedy steps ----------------
    for (int step = 0; step < STEPS; ++step) {
        int tok = sh_tok;
        sh_xbf[tid] = f32_to_bf16(dec_emb[tok * H + tid]);
        __syncthreads();

        {
            const __bf16* W   = (wave < 4) ? decWih  : decWhh;
            const __bf16* vec = (wave < 4) ? sh_xbf  : sh_hbf;
            float*        og  = (wave < 4) ? sh_gi   : sh_gh;
            matvec_tiles(W, vec, og, (wave & 3) * TILES_PER_WAVE,
                         TILES_PER_WAVE, lane);
        }
        __syncthreads();

        {
            int i = tid;
            float ir = sh_gi[i]         + dec_bih[i];
            float iz = sh_gi[H + i]     + dec_bih[H + i];
            float in = sh_gi[2 * H + i] + dec_bih[2 * H + i];
            float hr = sh_gh[i]         + dec_bhh[i];
            float hz = sh_gh[H + i]     + dec_bhh[H + i];
            float hn = sh_gh[2 * H + i] + dec_bhh[2 * H + i];
            float r  = sigmoidf_(ir + hr);
            float z  = sigmoidf_(iz + hz);
            float n  = tanhf(in + r * hn);
            float hv = (1.0f - z) * n + z * sh_h[i];
            sh_h[i]   = hv;
            sh_hbf[i] = f32_to_bf16(hv);
        }
        __syncthreads();

        // logits = fc_W @ h + fc_b : 5 padded tiles, waves 0..4 (uniform per wave)
        if (wave < FC_TILES)
            matvec_tiles(fcW, sh_hbf, sh_logits, wave, 1, lane);
        __syncthreads();

        if (tid == 0) {
            float best = -3.4e38f;
            int   bi   = 0;
            for (int o = 0; o < V_OUT; ++o) {
                float v = sh_logits[o] + fc_b[o];
                if (v > best) { best = v; bi = o; }   // strict '>' == first-max
            }
            int is_eos = (bi == EOS) ? 1 : 0;
            int pred   = (sh_done | is_eos) ? 0 : bi;
            out[step]  = (float)pred;
            sh_done   |= is_eos;
            sh_tok     = bi;
        }
        __syncthreads();
    }
}

// ---------------------------------------------------------------------------
extern "C" void kernel_launch(void* const* d_in, const int* in_sizes, int n_in,
                              void* d_out, int out_size, void* d_ws, size_t ws_size,
                              hipStream_t stream) {
    (void)in_sizes; (void)n_in; (void)out_size; (void)ws_size;

    const int*   enc_input = (const int*)  d_in[0];
    const float* enc_emb   = (const float*)d_in[1];
    const float* enc_Wih   = (const float*)d_in[2];
    const float* enc_Whh   = (const float*)d_in[3];
    const float* enc_bih   = (const float*)d_in[4];
    const float* enc_bhh   = (const float*)d_in[5];
    const float* dec_emb   = (const float*)d_in[6];
    const float* dec_Wih   = (const float*)d_in[7];
    const float* dec_Whh   = (const float*)d_in[8];
    const float* dec_bih   = (const float*)d_in[9];
    const float* dec_bhh   = (const float*)d_in[10];
    const float* fc_W      = (const float*)d_in[11];
    const float* fc_b      = (const float*)d_in[12];

    // workspace layout: bf16 fragment-major weights (~1.6 MB, stays in L2)
    const size_t GATE_ELEMS = (size_t)GATE_TILES * TILE_ELEMS;   // 196608
    const size_t FC_ELEMS   = (size_t)FC_TILES   * TILE_ELEMS;   // 20480
    __bf16* ws       = (__bf16*)d_ws;
    __bf16* encWih_f = ws;
    __bf16* encWhh_f = ws + 1 * GATE_ELEMS;
    __bf16* decWih_f = ws + 2 * GATE_ELEMS;
    __bf16* decWhh_f = ws + 3 * GATE_ELEMS;
    __bf16* fcW_f    = ws + 4 * GATE_ELEMS;

    const int gate_total  = (int)GATE_ELEMS;
    const int fc_total    = (int)FC_ELEMS;
    const int gate_blocks = (gate_total + 255) / 256;
    const int fc_blocks   = (fc_total + 255) / 256;

    convert_weights_k<<<gate_blocks, 256, 0, stream>>>(enc_Wih, encWih_f, 3 * H, gate_total);
    convert_weights_k<<<gate_blocks, 256, 0, stream>>>(enc_Whh, encWhh_f, 3 * H, gate_total);
    convert_weights_k<<<gate_blocks, 256, 0, stream>>>(dec_Wih, decWih_f, 3 * H, gate_total);
    convert_weights_k<<<gate_blocks, 256, 0, stream>>>(dec_Whh, decWhh_f, 3 * H, gate_total);
    convert_weights_k<<<fc_blocks,   256, 0, stream>>>(fc_W,    fcW_f,    V_OUT, fc_total);

    seq2seq_main_k<<<1, 256, 0, stream>>>(
        enc_input, enc_emb, enc_bih, enc_bhh,
        dec_emb, dec_bih, dec_bhh, fc_b,
        encWih_f, encWhh_f, decWih_f, decWhh_f, fcW_f,
        (float*)d_out);
}
